// HybridSelfAttention_65481071395957
// MI455X (gfx1250) — matrix operand order repaired and checked by reading, not compiled, not run on hardware
//
#include <hip/hip_runtime.h>
#include <hip/hip_bf16.h>

typedef __attribute__((ext_vector_type(16))) __bf16 v16bf;
typedef __attribute__((ext_vector_type(8)))  float  v8f;

__device__ __forceinline__ unsigned short f2bf(float f) {
    unsigned u = __float_as_uint(f);
    unsigned r = u + 0x7FFFu + ((u >> 16) & 1u);   // round-to-nearest-even
    return (unsigned short)(r >> 16);
}

// CDNA5 async copy: 16B/lane global -> LDS, tracked by ASYNCcnt (no VGPR staging)
__device__ __forceinline__ void async_cp16(const void* g, unsigned lds_off) {
    asm volatile("global_load_async_to_lds_b128 %0, %1, off"
                 :: "v"(lds_off), "v"(g) : "memory");
}
__device__ __forceinline__ void wait_async0() {
    asm volatile("s_wait_asynccnt 0x0" ::: "memory");
}
__device__ __forceinline__ unsigned lds_off_of(const void* p) {
    return (unsigned)(size_t)p;   // flat->LDS aperture: low 32 bits = LDS byte offset
}

// ---------------------------------------------------------------------------
// Convert x (fp32) -> bf16, vectorized 4-wide
// ---------------------------------------------------------------------------
__global__ void cvt_x_kernel(const float* __restrict__ x, unsigned short* __restrict__ y) {
    size_t i = ((size_t)blockIdx.x * 256 + threadIdx.x) * 4;
    float4 v = *(const float4*)(x + i);
    union { unsigned short h[4]; uint2 u; } pk;
    pk.h[0] = f2bf(v.x); pk.h[1] = f2bf(v.y); pk.h[2] = f2bf(v.z); pk.h[3] = f2bf(v.w);
    *(uint2*)(y + i) = pk.u;
}

// ---------------------------------------------------------------------------
// Convert + transpose weights: wt[z][n][k] = w_z[k][n]  (bf16), 32x32 LDS tiles
// ---------------------------------------------------------------------------
__global__ void cvt_w_kernel(const float* __restrict__ wq,
                             const float* __restrict__ wk,
                             const float* __restrict__ wv,
                             unsigned short* __restrict__ wt) {
    const float* w = (blockIdx.z == 0) ? wq : ((blockIdx.z == 1) ? wk : wv);
    unsigned short* dst = wt + (size_t)blockIdx.z * 1024 * 1024;
    __shared__ float tile[32][33];
    const int k0 = blockIdx.y * 32, n0 = blockIdx.x * 32;
    const int tx = threadIdx.x, ty = threadIdx.y;        // block = (32, 8)
    for (int i = 0; i < 32; i += 8)
        tile[ty + i][tx] = w[(size_t)(k0 + ty + i) * 1024 + (n0 + tx)];
    __syncthreads();
    for (int i = 0; i < 32; i += 8)
        dst[(size_t)(n0 + ty + i) * 1024 + (k0 + tx)] = f2bf(tile[tx][ty + i]);
}

// ---------------------------------------------------------------------------
// bf16 WMMA GEMM:  C[M,N] = A[M,K] * Bt[N,K]^T   (Bt is pre-transposed B)
// 128x128 block tile, BK=32, 8 waves (wave32), double-buffered LDS filled by
// CDNA5 async global->LDS copies (ASYNCcnt).
// mode 0: fused QKV epilogue (bf16 Q, K; V written transposed [b][e][s])
// mode 1: fp32 out with scale (attention scores)
// mode 2: fp32 out (final attention output)
// ---------------------------------------------------------------------------
#define BM 128
#define BN 128
#define BK 32
#define LDT 40   // LDS row stride in bf16 elements (80B: 16B-aligned, bank-spread)

__global__ __launch_bounds__(256) void wmma_gemm_kernel(
    const unsigned short* __restrict__ A,
    const unsigned short* __restrict__ Bt,
    int K, int N,
    unsigned short* __restrict__ oq,
    unsigned short* __restrict__ okk,
    unsigned short* __restrict__ ovt,
    float* __restrict__ of,
    float scale, int mode)
{
    __shared__ unsigned short As[2][BM][LDT];
    __shared__ unsigned short Bs[2][BM][LDT];

    const int tid  = threadIdx.x;
    const int lane = tid & 31;
    const int wave = tid >> 5;
    const int wm   = wave & 3;          // 4 waves tile M (4*32 = 128)
    const int wn   = wave >> 2;         // 2 waves tile N (2*64 = 128)
    const int blockM = blockIdx.y * BM;
    const int blockN = blockIdx.x * BN;

    // global->LDS copy role: each thread moves one 16B chunk per 64-row pass
    const int cpyC = (tid & 3) * 8;     // k offset in elements
    const int cpyR = tid >> 2;          // 0..63

    const int nK = K / BK;

    // per-thread source/dest bases
    const unsigned short* gA0 = A  + (size_t)(blockM + cpyR)      * K + cpyC;
    const unsigned short* gA1 = A  + (size_t)(blockM + cpyR + 64) * K + cpyC;
    const unsigned short* gB0 = Bt + (size_t)(blockN + cpyR)      * K + cpyC;
    const unsigned short* gB1 = Bt + (size_t)(blockN + cpyR + 64) * K + cpyC;
    const unsigned lA0[2] = { lds_off_of(&As[0][cpyR][cpyC]),      lds_off_of(&As[1][cpyR][cpyC]) };
    const unsigned lA1[2] = { lds_off_of(&As[0][cpyR + 64][cpyC]), lds_off_of(&As[1][cpyR + 64][cpyC]) };
    const unsigned lB0[2] = { lds_off_of(&Bs[0][cpyR][cpyC]),      lds_off_of(&Bs[1][cpyR][cpyC]) };
    const unsigned lB1[2] = { lds_off_of(&Bs[0][cpyR + 64][cpyC]), lds_off_of(&Bs[1][cpyR + 64][cpyC]) };

    // ---- prologue: async-stage k-tile 0 into buffer 0 ----
    async_cp16(gA0, lA0[0]);
    async_cp16(gA1, lA1[0]);
    async_cp16(gB0, lB0[0]);
    async_cp16(gB1, lB1[0]);
    wait_async0();
    __syncthreads();

    v8f acc[2][4];
    const v8f vzero = {0.f, 0.f, 0.f, 0.f, 0.f, 0.f, 0.f, 0.f};
    for (int i = 0; i < 2; ++i)
        for (int j = 0; j < 4; ++j) acc[i][j] = vzero;

    // per-lane fragment addressing (ISA 7.12.2 layouts, wave32)
    const int arow = lane & 15;               // A: row within 16x32 frag
    const int akb  = (lane >> 4) * 8;         // A: K base (0 or 8), second half at +16
    const int bkb  = (lane >> 4) * 8;         // B: K base (0 or 8), second half at +16

    for (int kt = 0; kt < nK; ++kt) {
        const int cur = kt & 1;
        const bool pre = (kt + 1) < nK;
        if (pre) {
            const int nxt = cur ^ 1;
            const int kO  = (kt + 1) * BK;
            async_cp16(gA0 + kO, lA0[nxt]);
            async_cp16(gA1 + kO, lA1[nxt]);
            async_cp16(gB0 + kO, lB0[nxt]);
            async_cp16(gB1 + kO, lB1[nxt]);
        }

        v16bf afrag[2], bfrag[4];
#pragma unroll
        for (int fm = 0; fm < 2; ++fm) {
            const unsigned short* p = &As[cur][wm * 32 + fm * 16 + arow][0];
            union { uint4 q[2]; v16bf v; } u;
            u.q[0] = *(const uint4*)(p + akb);
            u.q[1] = *(const uint4*)(p + akb + 16);
            afrag[fm] = u.v;
        }
#pragma unroll
        for (int fn = 0; fn < 4; ++fn) {
            const unsigned short* p = &Bs[cur][wn * 64 + fn * 16 + (lane & 15)][0];
            union { uint4 q[2]; v16bf v; } u;
            u.q[0] = *(const uint4*)(p + bkb);
            u.q[1] = *(const uint4*)(p + bkb + 16);
            bfrag[fn] = u.v;
        }

#pragma unroll
        for (int fm = 0; fm < 2; ++fm)
#pragma unroll
            for (int fn = 0; fn < 4; ++fn)
                acc[fm][fn] = __builtin_amdgcn_wmma_f32_16x16x32_bf16(
                    false, afrag[fm], false, bfrag[fn],
                    (short)0, acc[fm][fn], false, false);

        if (pre) wait_async0();
        __syncthreads();
    }

    // ---- epilogue (C/D layout: m = 8*(lane>>4)+r, n = lane&15 within frag) ----
    const int mrow0 = blockM + wm * 32 + (lane >> 4) * 8;
    const int ncol  = lane & 15;
#pragma unroll
    for (int fm = 0; fm < 2; ++fm) {
#pragma unroll
        for (int fn = 0; fn < 4; ++fn) {
#pragma unroll
            for (int r = 0; r < 8; ++r) {
                const int m = mrow0 + fm * 16 + r;
                const int n = blockN + wn * 64 + fn * 16 + ncol;
                const float val = acc[fm][fn][r];
                if (mode == 0) {
                    const unsigned short h = f2bf(val);
                    if (n < 1024) {
                        oq[(size_t)m * 1024 + n] = h;
                    } else if (n < 2048) {
                        okk[(size_t)m * 1024 + (n - 1024)] = h;
                    } else {
                        const int bb = m >> 11, s = m & 2047, e = n - 2048;
                        ovt[((size_t)bb * 1024 + e) * 2048 + s] = h;   // V transposed
                    }
                } else if (mode == 1) {
                    of[(size_t)m * N + n] = val * scale;
                } else {
                    of[(size_t)m * N + n] = val;
                }
            }
        }
    }
}

// ---------------------------------------------------------------------------
// Row softmax: S (fp32, ncols) -> P (bf16). One block per row, 256 threads.
// ---------------------------------------------------------------------------
__global__ void softmax_kernel(const float* __restrict__ S,
                               unsigned short* __restrict__ P, int ncols) {
    const int row = blockIdx.x;
    const float* srow = S + (size_t)row * ncols;
    __shared__ float red[256];
    const int tid = threadIdx.x;

    float m = -3.4e38f;
    for (int c = tid; c < ncols; c += 256) m = fmaxf(m, srow[c]);
    red[tid] = m; __syncthreads();
    for (int s = 128; s > 0; s >>= 1) {
        if (tid < s) red[tid] = fmaxf(red[tid], red[tid + s]);
        __syncthreads();
    }
    m = red[0]; __syncthreads();

    float sum = 0.f;
    for (int c = tid; c < ncols; c += 256) sum += __expf(srow[c] - m);
    red[tid] = sum; __syncthreads();
    for (int s = 128; s > 0; s >>= 1) {
        if (tid < s) red[tid] += red[tid + s];
        __syncthreads();
    }
    const float inv = 1.0f / red[0];

    unsigned short* prow = P + (size_t)row * ncols;
    for (int c = tid; c < ncols; c += 256)
        prow[c] = f2bf(__expf(srow[c] - m) * inv);
}

// ---------------------------------------------------------------------------
// Host launch
// ---------------------------------------------------------------------------
extern "C" void kernel_launch(void* const* d_in, const int* in_sizes, int n_in,
                              void* d_out, int out_size, void* d_ws, size_t ws_size,
                              hipStream_t stream) {
    const float* x  = (const float*)d_in[0];   // [4,2048,1024]
    const float* wq = (const float*)d_in[1];   // [1024,1024]
    const float* wk = (const float*)d_in[2];
    const float* wv = (const float*)d_in[3];
    float* out = (float*)d_out;                // [4,2048,1024] fp32

    char* ws = (char*)d_ws;
    const size_t MB = 1024 * 1024;
    unsigned short* xb = (unsigned short*)(ws);            // x bf16      [8192][1024]  16 MiB
    unsigned short* wt = (unsigned short*)(ws + 16 * MB);  // Wt bf16     [3][1024][1024] 6 MiB
    unsigned short* qb = (unsigned short*)(ws + 22 * MB);  // Q bf16      [8192][1024]  16 MiB
    unsigned short* kb = (unsigned short*)(ws + 38 * MB);  // K bf16      [8192][1024]  16 MiB
    unsigned short* vt = (unsigned short*)(ws + 54 * MB);  // V^T bf16    [4][1024][2048] 16 MiB
    float*          Sb = (float*)(ws + 70 * MB);           // scores fp32 [2048][2048]  16 MiB (reused per batch)
    unsigned short* Pb = (unsigned short*)(ws + 86 * MB);  // P bf16      [2048][2048]   8 MiB (reused per batch)

    // 1) precision conversion + weight transpose
    cvt_x_kernel<<<8192 * 1024 / (256 * 4), 256, 0, stream>>>(x, xb);
    cvt_w_kernel<<<dim3(32, 32, 3), dim3(32, 8), 0, stream>>>(wq, wk, wv, wt);

    // 2) fused QKV projection: [8192,1024] x [1024,3072]
    wmma_gemm_kernel<<<dim3(3072 / 128, 8192 / 128), 256, 0, stream>>>(
        xb, wt, 1024, 3072, qb, kb, vt, nullptr, 1.0f, 0);

    // 3) per-batch attention: scores -> softmax -> PV
    const float sc = 1.0f / 32.0f;   // 1/sqrt(1024)
    for (int b = 0; b < 4; ++b) {
        const unsigned short* Qb_ = qb + (size_t)b * 2048 * 1024;
        const unsigned short* Kb_ = kb + (size_t)b * 2048 * 1024;
        wmma_gemm_kernel<<<dim3(2048 / 128, 2048 / 128), 256, 0, stream>>>(
            Qb_, Kb_, 1024, 2048, nullptr, nullptr, nullptr, Sb, sc, 1);

        softmax_kernel<<<2048, 256, 0, stream>>>(Sb, Pb, 2048);

        wmma_gemm_kernel<<<dim3(1024 / 128, 2048 / 128), 256, 0, stream>>>(
            Pb, vt + (size_t)b * 1024 * 2048, 2048, 1024,
            nullptr, nullptr, nullptr, out + (size_t)b * 2048 * 1024, 1.0f, 2);
    }
}
